// CSFCN_12214886990335
// MI455X (gfx1250) — matrix operand order repaired and checked by reading, not compiled, not run on hardware
//
#include <hip/hip_runtime.h>
#include <cmath>

// ---------------------------------------------------------------------------
// MI455X (gfx1250) implementation.
// Conv layers: implicit-GEMM, bf16 WMMA (V_WMMA_F32_16X16X32_BF16, f32 acc),
// 256-thread / 8-wave workgroups, BMx128 tiles, double-buffered LDS software
// pipeline (global fetch of chunk k+1 overlaps WMMA on chunk k).
// Everything else (psp / attention / resize / grid-sample): bandwidth-bound.
// Scratch (floats), phase reuse; requires ws_size >= ~184 MB.
// ---------------------------------------------------------------------------

typedef __attribute__((ext_vector_type(16))) __bf16 v16bf;
typedef __attribute__((ext_vector_type(8)))  __bf16 v8bf;
typedef __attribute__((ext_vector_type(8)))  float  v8f;

#define LDS_STRIDE 40   // 32 K-values + 8 pad => 80B row stride, conflict-free b128 access

// ---------------------------------------------------------------------------
// Implicit-GEMM conv, stride 1, 'SAME' pad, NCHW/OIHW, fp32 in/out, bf16 WMMA.
// Virtual channel-concat: ci < C0 -> in0 else in1 (pass in1=in0, C0=Cin).
// Template: BM = 64 or 128 (M tile), KHC = kernel size (1 or 3, compile-time
// so im2col index math uses constant-division). BN fixed at 128.
// 8 waves: wave (r,c) = (wave>>2, wave&3); wave tile = (BM/2) x 32.
// Requires K % 32 == 0 (true for all convs here); N edge handled by guards.
// ---------------------------------------------------------------------------
template<int BM, int KHC>
__global__ __launch_bounds__(256)
void conv_igemm_wmma(const float* __restrict__ in0,
                     const float* __restrict__ in1,
                     int C0, int Cin,
                     const float* __restrict__ wgt,
                     const float* __restrict__ bias,
                     float* __restrict__ out,
                     int Cout, int H, int W, int act)
{
    constexpr int KK  = KHC * KHC;
    constexpr int pad = KHC >> 1;
    constexpr int BN  = 128;
    constexpr int MT  = BM / 32;          // 16x16 tiles per wave in M
    const int HW = H * W;
    const int K  = Cin * KK;
    const int b     = blockIdx.z;
    const int nBase = blockIdx.x * BN;
    const int cBase = blockIdx.y * BM;

    __shared__ __bf16 Asl[2][BM][LDS_STRIDE];
    __shared__ __bf16 Bsl[2][BN][LDS_STRIDE];

    const int tid  = threadIdx.x;
    const int lane = tid & 31;
    const int wave = tid >> 5;
    const int half = lane >> 4;
    const int l16  = lane & 15;
    const int wm   = (wave >> 2) * (BM / 2);
    const int wn   = (wave & 3) * 32;

    // staging roles: each thread owns one 16-K half-row of A (if arow<BM) + B
    const int arow  = tid >> 1;
    const int ahalf = tid & 1;
    const int brow  = tid >> 1;           // 0..127
    const int bhalf = tid & 1;
    const int npix  = nBase + brow;
    const int py    = npix / W;
    const int px    = npix - py * W;
    const bool nvalid = (npix < HW);
    const int aco   = cBase + arow;
    const bool avalid = (arow < BM) && (aco < Cout);
    const float* wrow = wgt + (size_t)aco * K + ahalf * 16;

    v8f acc[MT][2] = {};
    float aw[16];
    float bw[16];

    // fetch chunk [k0, k0+32) from global into registers
    auto fetch = [&](int k0) {
        if (avalid) {
            const float4* wp = reinterpret_cast<const float4*>(wrow + k0);
            __builtin_prefetch(wrow + k0 + 64, 0, 3);   // chunk after next in this row
#pragma unroll
            for (int j = 0; j < 4; ++j) {
                float4 v = wp[j];
                aw[j*4+0] = v.x; aw[j*4+1] = v.y; aw[j*4+2] = v.z; aw[j*4+3] = v.w;
            }
        }
#pragma unroll
        for (int j = 0; j < 16; ++j) {
            const int kg = k0 + bhalf * 16 + j;
            const int ci = kg / KK;                 // compile-time-constant divisor
            const int r  = kg - ci * KK;
            const int fy = r / KHC;
            const int fx = r - fy * KHC;
            const int yy = py + fy - pad;
            const int xx = px + fx - pad;
            float v = 0.0f;
            if (nvalid && yy >= 0 && yy < H && xx >= 0 && xx < W) {
                if (ci < C0)
                    v = in0[(((size_t)b * C0 + ci) * H + yy) * W + xx];
                else
                    v = in1[(((size_t)b * (Cin - C0) + (ci - C0)) * H + yy) * W + xx];
            }
            bw[j] = v;
        }
    };
    // convert + store registers into LDS buffer `buf` -- explicit 16B-aligned
    // v8bf stores so the compiler emits ds_store_b128 (dst offsets are 0/32/+16B
    // on an 80B row stride => always 16B aligned).
    auto commit = [&](int buf) {
        if (arow < BM) {
            v8bf pa0, pa1;
            if (avalid) {
#pragma unroll
                for (int j = 0; j < 8; ++j) { pa0[j] = (__bf16)aw[j]; pa1[j] = (__bf16)aw[8+j]; }
            } else {
#pragma unroll
                for (int j = 0; j < 8; ++j) { pa0[j] = (__bf16)0.0f; pa1[j] = (__bf16)0.0f; }
            }
            *reinterpret_cast<v8bf*>(&Asl[buf][arow][ahalf * 16])     = pa0;
            *reinterpret_cast<v8bf*>(&Asl[buf][arow][ahalf * 16 + 8]) = pa1;
        }
        v8bf pb0, pb1;
#pragma unroll
        for (int j = 0; j < 8; ++j) { pb0[j] = (__bf16)bw[j]; pb1[j] = (__bf16)bw[8+j]; }
        *reinterpret_cast<v8bf*>(&Bsl[buf][brow][bhalf * 16])     = pb0;
        *reinterpret_cast<v8bf*>(&Bsl[buf][brow][bhalf * 16 + 8]) = pb1;
    };

    const int nk = K >> 5;
    fetch(0);
    commit(0);

    for (int kc = 0; kc < nk; ++kc) {
        const int cur = kc & 1;
        if (kc + 1 < nk) fetch((kc + 1) << 5);      // global loads overlap WMMA below
        __syncthreads();                             // buf[cur] fully staged

        // fragments per ISA 16-bit A/B VGPR layouts (two b128 DS loads each)
        v16bf afrag[MT], bfrag[2];
#pragma unroll
        for (int t = 0; t < MT; ++t) {
            const int m = wm + t * 16 + l16;
            v8bf lo = *reinterpret_cast<const v8bf*>(&Asl[cur][m][half * 8]);
            v8bf hi = *reinterpret_cast<const v8bf*>(&Asl[cur][m][16 + half * 8]);
#pragma unroll
            for (int e = 0; e < 8; ++e) { afrag[t][e] = lo[e]; afrag[t][8+e] = hi[e]; }
        }
#pragma unroll
        for (int t = 0; t < 2; ++t) {
            const int n = wn + t * 16 + l16;
            v8bf lo = *reinterpret_cast<const v8bf*>(&Bsl[cur][n][half * 8]);
            v8bf hi = *reinterpret_cast<const v8bf*>(&Bsl[cur][n][16 + half * 8]);
#pragma unroll
            for (int e = 0; e < 8; ++e) { bfrag[t][e] = lo[e]; bfrag[t][8+e] = hi[e]; }
        }

#pragma unroll
        for (int tm = 0; tm < MT; ++tm)
#pragma unroll
            for (int tn = 0; tn < 2; ++tn)
                acc[tm][tn] = __builtin_amdgcn_wmma_f32_16x16x32_bf16(
                    false, afrag[tm], false, bfrag[tn],
                    (short)0, acc[tm][tn], false, false);

        if (kc + 1 < nk) commit((kc + 1) & 1);       // DS stores after frag reads
    }

    // epilogue: C/D layout = (M = r + 8*half, N = lane&15)
#pragma unroll
    for (int tm = 0; tm < MT; ++tm) {
#pragma unroll
        for (int tn = 0; tn < 2; ++tn) {
#pragma unroll
            for (int r = 0; r < 8; ++r) {
                const int co = cBase + wm + tm * 16 + r + 8 * half;
                const int np = nBase + wn + tn * 16 + l16;
                if (co < Cout && np < HW) {
                    float v = acc[tm][tn][r];
                    if (bias) v += bias[co];
                    if (act == 1) v = v / (1.0f + __expf(-v));   // SiLU
                    out[(((size_t)b * Cout + co) * HW) + np] = v;
                }
            }
        }
    }
}

// ---------------------------------------------------------------------------
// PSP: adaptive avg pool to grids (6,3,2,1) -> kfeat [B,C,50]
// ---------------------------------------------------------------------------
__global__ void psp_kernel(const float* __restrict__ x, float* __restrict__ kf,
                           int B_, int C, int H, int W)
{
    int idx = blockIdx.x * blockDim.x + threadIdx.x;
    int total = B_ * C * 50;
    if (idx >= total) return;
    int m = idx % 50;
    int c = (idx / 50) % C;
    int b = idx / (50 * C);
    int g, base;
    if (m < 36)      { g = 6; base = 0;  }
    else if (m < 45) { g = 3; base = 36; }
    else if (m < 49) { g = 2; base = 45; }
    else             { g = 1; base = 49; }
    int loc = m - base;
    int gy = loc / g, gx = loc % g;
    int bh = H / g, bw = W / g;
    const float* src = x + ((size_t)b * C + c) * H * W;
    float s = 0.f;
    for (int yy = gy * bh; yy < (gy + 1) * bh; ++yy)
        for (int xx = gx * bw; xx < (gx + 1) * bw; ++xx)
            s += src[yy * W + xx];
    kf[((size_t)b * C + c) * 50 + m] = s / (float)(bh * bw);
}

// out[b,o,m] = sum_c Wm[o,c]*kf[b,c,m] + bias[o]
__global__ void linear50_kernel(const float* __restrict__ Wm, const float* __restrict__ bias,
                                const float* __restrict__ kf, float* __restrict__ out,
                                int B_, int O, int C)
{
    int idx = blockIdx.x * blockDim.x + threadIdx.x;
    int total = B_ * O * 50;
    if (idx >= total) return;
    int m = idx % 50;
    int o = (idx / 50) % O;
    int b = idx / (50 * O);
    float s = bias[o];
    const float* wrow = Wm + (size_t)o * C;
    const float* kcol = kf + (size_t)b * C * 50 + m;
    for (int c = 0; c < C; ++c) s += wrow[c] * kcol[(size_t)c * 50];
    out[((size_t)b * O + o) * 50 + m] = s;
}

// sim[b,n,:] = softmax_m( q[b,:,n] . kmat[b,:,m] )
__global__ void attn_softmax_kernel(const float* __restrict__ q, const float* __restrict__ kmat,
                                    float* __restrict__ sim, int B_, int N)
{
    int idx = blockIdx.x * blockDim.x + threadIdx.x;
    if (idx >= B_ * N) return;
    int n = idx % N;
    int b = idx / N;
    float logits[50];
#pragma unroll
    for (int m = 0; m < 50; ++m) logits[m] = 0.f;
    for (int c = 0; c < 32; ++c) {
        float qc = q[((size_t)b * 32 + c) * N + n];
        const float* krow = kmat + ((size_t)b * 32 + c) * 50;
#pragma unroll
        for (int m = 0; m < 50; ++m) logits[m] += qc * krow[m];
    }
    float mx = logits[0];
#pragma unroll
    for (int m = 1; m < 50; ++m) mx = fmaxf(mx, logits[m]);
    float den = 0.f;
#pragma unroll
    for (int m = 0; m < 50; ++m) { logits[m] = __expf(logits[m] - mx); den += logits[m]; }
    float inv = 1.f / den;
    float* dst = sim + ((size_t)b * N + n) * 50;
#pragma unroll
    for (int m = 0; m < 50; ++m) dst[m] = logits[m] * inv;
}

// ctx[b,c,n] = sum_m v[b,c,m]*sim[b,n,m]
__global__ void context_kernel(const float* __restrict__ vmat, const float* __restrict__ sim,
                               float* __restrict__ ctx, int B_, int C, int N)
{
    int idx = blockIdx.x * blockDim.x + threadIdx.x;
    int total = B_ * C * N;
    if (idx >= total) return;
    int n = idx % N;
    int c = (idx / N) % C;
    int b = idx / (N * C);
    const float* vrow = vmat + ((size_t)b * C + c) * 50;
    const float* srow = sim + ((size_t)b * N + n) * 50;
    float s = 0.f;
#pragma unroll
    for (int m = 0; m < 50; ++m) s += vrow[m] * srow[m];
    ctx[((size_t)b * C + c) * N + n] = s;
}

// p3o = x + ctx*tanh(t2) + ctx
__global__ void p3o_kernel(const float* __restrict__ x, const float* __restrict__ ctx,
                           const float* __restrict__ t2, float* __restrict__ outp, int n)
{
    int i = blockIdx.x * blockDim.x + threadIdx.x;
    if (i >= n) return;
    float c = ctx[i];
    outp[i] = x[i] + c * tanhf(t2[i]) + c;
}

// bilinear resize align_corners=True
__global__ void resize_kernel(const float* __restrict__ src, float* __restrict__ dst,
                              int B_, int C, int h, int w, int oh, int ow)
{
    int idx = blockIdx.x * blockDim.x + threadIdx.x;
    int total = B_ * C * oh * ow;
    if (idx >= total) return;
    int x = idx % ow;
    int y = (idx / ow) % oh;
    int c = (idx / (ow * oh)) % C;
    int b = idx / (ow * oh * C);
    float sy = (float)y * (float)(h - 1) / (float)(oh - 1);
    float sx = (float)x * (float)(w - 1) / (float)(ow - 1);
    int y0 = (int)floorf(sy); y0 = min(max(y0, 0), h - 2);
    int x0 = (int)floorf(sx); x0 = min(max(x0, 0), w - 2);
    float wy = sy - (float)y0, wx = sx - (float)x0;
    const float* s = src + (((size_t)b * C + c) * h) * w;
    float v00 = s[y0 * w + x0],     v01 = s[y0 * w + x0 + 1];
    float v10 = s[(y0+1) * w + x0], v11 = s[(y0+1) * w + x0 + 1];
    dst[idx] = (v00 * (1.f - wx) + v01 * wx) * (1.f - wy)
             + (v10 * (1.f - wx) + v11 * wx) * wy;
}

__device__ inline float gsample(const float* __restrict__ img, int b, int C, int c,
                                int H, int W, float gx, float gy)
{
    float px = (gx + 1.0f) * 0.5f * (float)(W - 1);
    float py = (gy + 1.0f) * 0.5f * (float)(H - 1);
    float x0f = floorf(px), y0f = floorf(py);
    int x0 = (int)x0f, y0 = (int)y0f;
    float wx = px - x0f, wy = py - y0f;
    const float* base = img + (((size_t)b * C + c) * H) * W;
    float s = 0.f;
#pragma unroll
    for (int dy = 0; dy < 2; ++dy)
#pragma unroll
        for (int dx = 0; dx < 2; ++dx) {
            int xi = x0 + dx, yi = y0 + dy;
            if (xi >= 0 && xi < W && yi >= 0 && yi < H) {
                float wt = (dx ? wx : 1.f - wx) * (dy ? wy : 1.f - wy);
                s += wt * base[yi * W + xi];
            }
        }
    return s;
}

// dual grid-sample + attention blend -> p5o
__global__ void sfc_out_kernel(const float* __restrict__ sp, const float* __restrict__ cp,
                               const float* __restrict__ cr, float* __restrict__ outp,
                               int B_, int C, int H, int W)
{
    int idx = blockIdx.x * blockDim.x + threadIdx.x;
    int total = B_ * C * H * W;
    if (idx >= total) return;
    int x = idx % W;
    int y = (idx / W) % H;
    int c = (idx / (W * H)) % C;
    int b = idx / (W * H * C);
    int g = (c * 2) / C;               // group of 128 channels
    int p = y * W + x;
    const size_t crb = (size_t)b * 10 * H * W;
    float offlx = cr[crb + (size_t)(2 * g + 0) * H * W + p];
    float offly = cr[crb + (size_t)(2 * g + 1) * H * W + p];
    float offhx = cr[crb + (size_t)(4 + 2 * g) * H * W + p];
    float offhy = cr[crb + (size_t)(5 + 2 * g) * H * W + p];
    float Xn = -1.f + 2.f * (float)x / (float)(W - 1);
    float Yn = -1.f + 2.f * (float)y / (float)(H - 1);
    float cps = gsample(cp, b, C, c, H, W, Xn + offlx / (float)W, Yn + offly / (float)H);
    float sps = gsample(sp, b, C, c, H, W, Xn + offhx / (float)W, Yn + offhy / (float)H);
    float att0 = 1.f + tanhf(cr[crb + (size_t)8 * H * W + p]);
    float att1 = 1.f + tanhf(cr[crb + (size_t)9 * H * W + p]);
    outp[idx] = sps * att0 + cps * att1;
}

// ---------------------------------------------------------------------------
extern "C" void kernel_launch(void* const* d_in, const int* in_sizes, int n_in,
                              void* d_out, int out_size, void* d_ws, size_t ws_size,
                              hipStream_t stream)
{
    const float* p3    = (const float*)d_in[0];
    const float* p5    = (const float*)d_in[1];
    const float* w_red = (const float*)d_in[2];
    const float* wq    = (const float*)d_in[3];
    const float* bq    = (const float*)d_in[4];
    const float* wk    = (const float*)d_in[5];
    const float* bk    = (const float*)d_in[6];
    const float* wv    = (const float*)d_in[7];
    const float* bv    = (const float*)d_in[8];
    const float* la_w1 = (const float*)d_in[9];
    const float* la_w2 = (const float*)d_in[10];
    const float* w32   = (const float*)d_in[11];
    const float* w8    = (const float*)d_in[12];
    const float* woff1 = (const float*)d_in[13];
    const float* woff2 = (const float*)d_in[14];
    float* out = (float*)d_out;
    float* ws  = (float*)d_ws;

    const int B_ = 8, C3 = 512, C5 = 256;
    const int N24 = 24 * 24;        // 576
    const int N96 = 96 * 96;        // 9216
    const int T24 = (N24 + 127) / 128;   // 5 (N-edge guarded)
    const int T96 = N96 / 128;           // 72
    const size_t P3O_SZ = (size_t)B_ * C3 * N24;   // 2359296

    // scratch layout (floats) with phase reuse
    float* s_x    = ws + 0;                       // [8,512,24,24]
    float* s_ctx  = ws + 2359296;                 // [8,512,24,24]
    float* s_t2   = ws + 4718592;                 // [8,512,24,24]
    float* s_kf   = ws + 7077888;                 // [8,512,50]
    float* s_km   = ws + 7282688;                 // [8,32,50]
    float* s_vm   = ws + 7295488;                 // [8,512,50]
    float* s_qm   = ws + 7500288;                 // [8,32,24,24]
    float* s_sim  = ws + 7647744;                 // [8,576,50]
    float* s_t1   = ws + 7878144;                 // [8,64,24,24]
    float* s_mid  = ws + 0;                       // [8,64,96,96]  (reuses x+ctx)
    float* s_sp24 = ws + 4718592;                 // [8,256,24,24] (reuses t2)
    float* s_cr   = ws + 7077888;                 // [8,10,96,96]  (reuses smalls)
    float* s_sp   = ws + 8173056;                 // [8,256,96,96]
    float* s_cp   = ws + 27047424;                // [8,256,96,96]

    const dim3 blk(256);
    #define G1D(total) dim3(((total) + 255) / 256), dim3(256)

    // ------------------- CFC_CRB -------------------
    // x = silu(conv3x3(p3, w_red))
    conv_igemm_wmma<128,3><<<dim3(T24, C3/128, B_), blk, 0, stream>>>(
        p3, p3, C3, C3, w_red, nullptr, s_x, C3, 24, 24, 1);
    // q = conv1x1(x, wq) + bq
    conv_igemm_wmma<64,1><<<dim3(T24, 1, B_), blk, 0, stream>>>(
        s_x, s_x, C3, C3, wq, bq, s_qm, 32, 24, 24, 0);
    // kfeat = psp(x)
    psp_kernel<<<G1D(B_*C3*50), 0, stream>>>(s_x, s_kf, B_, C3, 24, 24);
    // k = wk@kfeat + bk ; v = wv@kfeat + bv
    linear50_kernel<<<G1D(B_*32*50),  0, stream>>>(wk, bk, s_kf, s_km, B_, 32,  C3);
    linear50_kernel<<<G1D(B_*C3*50),  0, stream>>>(wv, bv, s_kf, s_vm, B_, C3, C3);
    // sim = softmax(q^T k)
    attn_softmax_kernel<<<G1D(B_*N24), 0, stream>>>(s_qm, s_km, s_sim, B_, N24);
    // context = v @ sim^T
    context_kernel<<<G1D(B_*C3*N24), 0, stream>>>(s_vm, s_sim, s_ctx, B_, C3, N24);
    // LocalAtten: t1 = silu(conv1x1(ctx, la_w1)); t2 = conv3x3(t1, la_w2)
    conv_igemm_wmma<64,1><<<dim3(T24, 1, B_), blk, 0, stream>>>(
        s_ctx, s_ctx, C3, C3, la_w1, nullptr, s_t1, 64, 24, 24, 1);
    conv_igemm_wmma<128,3><<<dim3(T24, C3/128, B_), blk, 0, stream>>>(
        s_t1, s_t1, 64, 64, la_w2, nullptr, s_t2, C3, 24, 24, 0);
    // p3o = x + ctx*tanh(t2) + ctx  -> d_out[0:P3O_SZ]
    p3o_kernel<<<G1D((int)P3O_SZ), 0, stream>>>(s_x, s_ctx, s_t2, out, (int)P3O_SZ);

    // ------------------- SFC_G2 -------------------
    // sp24 = silu(conv3x3(p3o, w32)); bilinear up to 96x96
    conv_igemm_wmma<128,3><<<dim3(T24, C5/128, B_), blk, 0, stream>>>(
        out, out, C3, C3, w32, nullptr, s_sp24, C5, 24, 24, 1);
    resize_kernel<<<G1D(B_*C5*N96), 0, stream>>>(s_sp24, s_sp, B_, C5, 24, 24, 96, 96);
    // cp = silu(conv3x3(p5, w8))  -- the big one: M=256, N=9216, K=2304
    conv_igemm_wmma<128,3><<<dim3(T96, C5/128, B_), blk, 0, stream>>>(
        p5, p5, C5, C5, w8, nullptr, s_cp, C5, 96, 96, 1);
    // mid = silu(conv1x1(concat[cp,sp], woff1)) -- virtual concat via C0 split
    conv_igemm_wmma<64,1><<<dim3(T96, 1, B_), blk, 0, stream>>>(
        s_cp, s_sp, C5, 2*C5, woff1, nullptr, s_mid, 64, 96, 96, 1);
    // cr = conv3x3(mid, woff2), Cout=10
    conv_igemm_wmma<64,3><<<dim3(T96, 1, B_), blk, 0, stream>>>(
        s_mid, s_mid, 64, 64, woff2, nullptr, s_cr, 10, 96, 96, 0);
    // p5o = grid_sample blend -> d_out[P3O_SZ:]
    sfc_out_kernel<<<G1D(B_*C5*N96), 0, stream>>>(s_sp, s_cp, s_cr, out + P3O_SZ,
                                                  B_, C5, 96, 96);
    #undef G1D
    (void)in_sizes; (void)n_in; (void)out_size; (void)ws_size;
}
